// DenseGIN_12352325943891
// MI455X (gfx1250) — matrix operand order repaired and checked by reading, not compiled, not run on hardware
//
#include <hip/hip_runtime.h>
#include <hip/hip_bf16.h>

// ---------------------------------------------------------------------------
// DenseGIN forward on gfx1250 (MI455X), bf16 WMMA (v_wmma_f32_16x16x32_bf16)
// B=4, N=4096, K=2; layer dims (32->64->64), (64->64->64), (64->64->32)
//
// adj GEMM: each wave owns a 16-row strip x ALL KC features -> adj (the 768MB
// HBM stream) is loaded exactly once. K-dim split 2-way across waves in a
// block, partials combined through LDS (deterministic, no atomics).
// ---------------------------------------------------------------------------

typedef __attribute__((ext_vector_type(16))) __bf16 bf16x16;
typedef __attribute__((ext_vector_type(8)))  float  f32x8;

#define GIN_N  4096
#define GIN_B  4
#define GIN_K  2

__device__ __forceinline__ unsigned short bf16_bits(float f) {
    union { __bf16 h; unsigned short s; } u;
    u.h = (__bf16)f;
    return u.s;
}

// ---------------------------------------------------------------------------
// Weight transpose + fp32 -> bf16:  W[R][C] -> WT[C][R]
// ---------------------------------------------------------------------------
__global__ void w_to_bf16T(const float* __restrict__ W,
                           unsigned short* __restrict__ WT,
                           int R, int C) {
    int idx = blockIdx.x * blockDim.x + threadIdx.x;
    if (idx >= R * C) return;
    int col = idx % C, row = idx / C;
    WT[col * R + row] = bf16_bits(W[idx]);
}

// ---------------------------------------------------------------------------
// Per-layer prep: x -> (optional outer-BN + mask) -> xf (f32, [B,N,KC]) and
// xfT (bf16, [B,KC,N]) so the GEMM B-operand is K-contiguous.
// ---------------------------------------------------------------------------
__global__ void gin_prep(const float* __restrict__ xin,
                         const unsigned char* __restrict__ mask,
                         const float* __restrict__ og, const float* __restrict__ ob,
                         const float* __restrict__ om, const float* __restrict__ ov,
                         int C, int applyBN,
                         float* __restrict__ xf,
                         unsigned short* __restrict__ xfT) {
    const int KC = GIN_K * C;
    size_t idx = (size_t)blockIdx.x * blockDim.x + threadIdx.x;
    size_t total = (size_t)GIN_B * GIN_N * KC;
    if (idx >= total) return;
    int    f  = (int)(idx % KC);         // k*C + c
    size_t bn = idx / KC;                // b*N + n
    int    c  = f % C;
    float v = xin[idx];
    if (applyBN) {
        float xn = (v - om[c]) * __frsqrt_rn(ov[c] + 1e-5f) * og[c] + ob[c];
        v = mask[bn] ? xn : 0.0f;
    }
    xf[idx] = v;
    int b = (int)(bn / GIN_N), n = (int)(bn % GIN_N);
    xfT[((size_t)b * KC + f) * GIN_N + n] = bf16_bits(v);
}

// ---------------------------------------------------------------------------
// agg[b,i,f] = sum_j adj[b,i,j] * xf[b,j,f] + xf[b,i,f]
// grid = (N/64, B), block = 256 (8 waves).
// wave = strip(0..3) + 4*khalf : strip owns rows i0+strip*16..+15 for ALL KC
// features; khalf selects j in [0,2048) or [2048,4096). khalf=1 partials are
// reduced into khalf=0 accumulators through LDS.
// ---------------------------------------------------------------------------
template<int KC>
__global__ __launch_bounds__(256) void gin_adj_gemm(
        const float* __restrict__ adj,
        const unsigned short* __restrict__ xfT,
        const float* __restrict__ xf,
        float* __restrict__ agg) {
    constexpr int N  = GIN_N;
    constexpr int NT = KC / 16;                        // 4 or 8 feature tiles
    __shared__ __align__(16) float partials[4][NT * 8 * 32];

    int wave  = threadIdx.x >> 5;
    int lane  = threadIdx.x & 31;
    int strip = wave & 3;
    int khalf = wave >> 2;
    int mrow  = lane & 15;
    int hi    = lane >> 4;
    int sA    = hi ? 8 : 0;    // 16-bit A layout: lanes 0-15 K base 0, 16-31 base 8
    int sB    = hi ? 8 : 0;    // 16-bit B layout: lanes 0-15 K base 0, 16-31 base 8

    int b  = blockIdx.y;
    int i0 = blockIdx.x * 64 + strip * 16;

    const float*          arow  = adj + ((size_t)b * N + (i0 + mrow)) * N;
    const unsigned short* bbase = xfT + (size_t)b * KC * N + (size_t)mrow * N + sB;

    int j0 = khalf * (N / 2);
    int j1 = j0 + (N / 2);

    f32x8 z = {0.f, 0.f, 0.f, 0.f, 0.f, 0.f, 0.f, 0.f};
    f32x8 acc[NT];
#pragma unroll
    for (int nt = 0; nt < NT; ++nt) acc[nt] = z;

#pragma unroll 2
    for (int j = j0; j < j1; j += 32) {
        if ((j & 1023) == 0) __builtin_prefetch(arow + j + 2048, 0, 0);
        // ---- A tile (16x32 bf16) from fp32 adj: unique per wave ----
        float4 p0 = *(const float4*)(arow + j + sA);
        float4 p1 = *(const float4*)(arow + j + sA + 4);
        float4 p2 = *(const float4*)(arow + j + sA + 16);
        float4 p3 = *(const float4*)(arow + j + sA + 20);
        bf16x16 a;
        a[0]  = (__bf16)p0.x; a[1]  = (__bf16)p0.y; a[2]  = (__bf16)p0.z; a[3]  = (__bf16)p0.w;
        a[4]  = (__bf16)p1.x; a[5]  = (__bf16)p1.y; a[6]  = (__bf16)p1.z; a[7]  = (__bf16)p1.w;
        a[8]  = (__bf16)p2.x; a[9]  = (__bf16)p2.y; a[10] = (__bf16)p2.z; a[11] = (__bf16)p2.w;
        a[12] = (__bf16)p3.x; a[13] = (__bf16)p3.y; a[14] = (__bf16)p3.z; a[15] = (__bf16)p3.w;
        // ---- B tiles (32x16 bf16) from xfT: tiny, L2/WGP$-resident ----
#pragma unroll
        for (int nt = 0; nt < NT; ++nt) {
            union { bf16x16 v; uint4 q[2]; } bu;
            const uint4* bp = (const uint4*)(bbase + (size_t)nt * 16 * N + j);
            bu.q[0] = bp[0];
            bu.q[1] = bp[2];
            acc[nt] = __builtin_amdgcn_wmma_f32_16x16x32_bf16(false, a, false, bu.v,
                                                              (short)0, acc[nt], false, false);
        }
    }

    // ---- combine K-halves through LDS ----
    if (khalf) {
#pragma unroll
        for (int nt = 0; nt < NT; ++nt)
#pragma unroll
            for (int r = 0; r < 8; ++r)
                partials[strip][(nt * 8 + r) * 32 + lane] = acc[nt][r];
    }
    __syncthreads();
    if (!khalf) {
        int fr0 = mrow;
#pragma unroll
        for (int nt = 0; nt < NT; ++nt) {
            int f = nt * 16 + fr0;
#pragma unroll
            for (int r = 0; r < 8; ++r) {
                float sum = acc[nt][r] + partials[strip][(nt * 8 + r) * 32 + lane];
                int row = i0 + r + hi * 8;
                size_t idx = ((size_t)b * N + row) * KC + f;
                agg[idx] = sum + xf[idx];        // GIN self term (eps = 0)
            }
        }
    }
}

// ---------------------------------------------------------------------------
// Fused node MLP:  h = BN(relu(agg@Wa + ba));  out = mask * (h@Wb + bb)
// rows = B*N*K = 32768 -> 2048 tiles of 16 rows; 8 waves / block.
// h re-staged through LDS to build the GEMM2 A operand.
// ---------------------------------------------------------------------------
template<int CIN, int COUT>
__global__ __launch_bounds__(256) void gin_mlp(
        const float* __restrict__ agg,                 // [rows, CIN]
        const unsigned short* __restrict__ WaT,        // [64, CIN] bf16
        const float* __restrict__ ba,
        const float* __restrict__ bng, const float* __restrict__ bnb,
        const float* __restrict__ bnm, const float* __restrict__ bnv,
        const unsigned short* __restrict__ WbT,        // [COUT, 64] bf16
        const float* __restrict__ bb,
        const unsigned char* __restrict__ mask,        // [B*N]
        float* __restrict__ out) {                     // [rows, COUT]
    __shared__ __align__(16) unsigned short hbuf[8][16 * 64];

    int wave = threadIdx.x >> 5;
    int lane = threadIdx.x & 31;
    int mrow = lane & 15;
    int hi   = lane >> 4;
    int sA   = hi ? 8 : 0;
    int sB   = hi ? 8 : 0;
    int r0   = (blockIdx.x * 8 + wave) * 16;

    // ---------------- GEMM1: agg(rows x CIN) @ Wa(CIN x 64) ----------------
    f32x8 z = {0.f, 0.f, 0.f, 0.f, 0.f, 0.f, 0.f, 0.f};
    f32x8 acc[4] = {z, z, z, z};
    const float* arow = agg + (size_t)(r0 + mrow) * CIN;

#pragma unroll
    for (int j = 0; j < CIN; j += 32) {
        float4 p0 = *(const float4*)(arow + j + sA);
        float4 p1 = *(const float4*)(arow + j + sA + 4);
        float4 p2 = *(const float4*)(arow + j + sA + 16);
        float4 p3 = *(const float4*)(arow + j + sA + 20);
        bf16x16 a;
        a[0]  = (__bf16)p0.x; a[1]  = (__bf16)p0.y; a[2]  = (__bf16)p0.z; a[3]  = (__bf16)p0.w;
        a[4]  = (__bf16)p1.x; a[5]  = (__bf16)p1.y; a[6]  = (__bf16)p1.z; a[7]  = (__bf16)p1.w;
        a[8]  = (__bf16)p2.x; a[9]  = (__bf16)p2.y; a[10] = (__bf16)p2.z; a[11] = (__bf16)p2.w;
        a[12] = (__bf16)p3.x; a[13] = (__bf16)p3.y; a[14] = (__bf16)p3.z; a[15] = (__bf16)p3.w;
#pragma unroll
        for (int nt = 0; nt < 4; ++nt) {
            union { bf16x16 v; uint4 q[2]; } bu;
            const uint4* bp = (const uint4*)(WaT + (size_t)(nt * 16 + mrow) * CIN + j + sB);
            bu.q[0] = bp[0];
            bu.q[1] = bp[2];
            acc[nt] = __builtin_amdgcn_wmma_f32_16x16x32_bf16(false, a, false, bu.v,
                                                              (short)0, acc[nt], false, false);
        }
    }

    // ---------------- epilogue1: +ba, relu, BN -> LDS (bf16) ----------------
#pragma unroll
    for (int nt = 0; nt < 4; ++nt) {
        int f = nt * 16 + mrow;
        float g  = bng[f], bt = bnb[f], m = bnm[f];
        float rs = __frsqrt_rn(bnv[f] + 1e-5f);
        float bav = ba[f];
#pragma unroll
        for (int r = 0; r < 8; ++r) {
            float x = acc[nt][r] + bav;
            x = x > 0.f ? x : 0.f;
            x = (x - m) * rs * g + bt;
            hbuf[wave][(r + hi * 8) * 64 + f] = bf16_bits(x);
        }
    }
    __syncthreads();

    // ---------------- GEMM2: h(rows x 64) @ Wb(64 x COUT) ------------------
    constexpr int NT2 = COUT / 16;
    f32x8 acc2[NT2];
#pragma unroll
    for (int nt = 0; nt < NT2; ++nt) acc2[nt] = z;

    const unsigned short* hrow = &hbuf[wave][mrow * 64];
#pragma unroll
    for (int j = 0; j < 64; j += 32) {
        union { bf16x16 v; uint4 q[2]; } au;
        au.q[0] = *(const uint4*)(hrow + j + sA);
        au.q[1] = *(const uint4*)(hrow + j + sA + 16);
#pragma unroll
        for (int nt = 0; nt < NT2; ++nt) {
            union { bf16x16 v; uint4 q[2]; } bu;
            const uint4* bp = (const uint4*)(WbT + (size_t)(nt * 16 + mrow) * 64 + j + sB);
            bu.q[0] = bp[0];
            bu.q[1] = bp[2];
            acc2[nt] = __builtin_amdgcn_wmma_f32_16x16x32_bf16(false, au.v, false, bu.v,
                                                               (short)0, acc2[nt], false, false);
        }
    }

    // ---------------- epilogue2: +bb, mask, store ---------------------------
#pragma unroll
    for (int nt = 0; nt < NT2; ++nt) {
        int f = nt * 16 + mrow;
        float bbv = bb[f];
#pragma unroll
        for (int r = 0; r < 8; ++r) {
            int R = r0 + r + hi * 8;
            float v = acc2[nt][r] + bbv;
            v = mask[R >> 1] ? v : 0.0f;     // K==2: node index = R/2
            out[(size_t)R * COUT + f] = v;
        }
    }
}

// ---------------------------------------------------------------------------
extern "C" void kernel_launch(void* const* d_in, const int* in_sizes, int n_in,
                              void* d_out, int out_size, void* d_ws, size_t ws_size,
                              hipStream_t stream) {
    (void)in_sizes; (void)n_in; (void)out_size; (void)ws_size;

    const float*         x    = (const float*)d_in[0];
    const float*         adj  = (const float*)d_in[1];
    const unsigned char* mask = (const unsigned char*)d_in[2];
    const float* Wa[3]  = {(const float*)d_in[3],  (const float*)d_in[11], (const float*)d_in[19]};
    const float* ba[3]  = {(const float*)d_in[4],  (const float*)d_in[12], (const float*)d_in[20]};
    const float* bng[3] = {(const float*)d_in[5],  (const float*)d_in[13], (const float*)d_in[21]};
    const float* bnb[3] = {(const float*)d_in[6],  (const float*)d_in[14], (const float*)d_in[22]};
    const float* bnm[3] = {(const float*)d_in[7],  (const float*)d_in[15], (const float*)d_in[23]};
    const float* bnv[3] = {(const float*)d_in[8],  (const float*)d_in[16], (const float*)d_in[24]};
    const float* Wb[3]  = {(const float*)d_in[9],  (const float*)d_in[17], (const float*)d_in[25]};
    const float* bb[3]  = {(const float*)d_in[10], (const float*)d_in[18], (const float*)d_in[26]};
    const float* og[2]  = {(const float*)d_in[27], (const float*)d_in[31]};
    const float* ob[2]  = {(const float*)d_in[28], (const float*)d_in[32]};
    const float* om[2]  = {(const float*)d_in[29], (const float*)d_in[33]};
    const float* ov[2]  = {(const float*)d_in[30], (const float*)d_in[34]};

    char* ws = (char*)d_ws;
    float*          xf   = (float*)(ws);                        // 8 MB
    float*          agg  = (float*)(ws + ((size_t)8  << 20));   // 8 MB
    float*          xcur = (float*)(ws + ((size_t)16 << 20));   // 8 MB
    unsigned short* xfT  = (unsigned short*)(ws + ((size_t)24 << 20)); // 4 MB
    size_t off = (size_t)28 << 20;
    unsigned short* WaT[3]; unsigned short* WbT[3];
    WaT[0] = (unsigned short*)(ws + off); off += 64 * 32 * 2;
    WaT[1] = (unsigned short*)(ws + off); off += 64 * 64 * 2;
    WaT[2] = (unsigned short*)(ws + off); off += 64 * 64 * 2;
    WbT[0] = (unsigned short*)(ws + off); off += 64 * 64 * 2;
    WbT[1] = (unsigned short*)(ws + off); off += 64 * 64 * 2;
    WbT[2] = (unsigned short*)(ws + off); off += 32 * 64 * 2;

    // Weight transposes -> bf16
    w_to_bf16T<<<(32 * 64 + 255) / 256, 256, 0, stream>>>(Wa[0], WaT[0], 32, 64);
    w_to_bf16T<<<(64 * 64 + 255) / 256, 256, 0, stream>>>(Wa[1], WaT[1], 64, 64);
    w_to_bf16T<<<(64 * 64 + 255) / 256, 256, 0, stream>>>(Wa[2], WaT[2], 64, 64);
    w_to_bf16T<<<(64 * 64 + 255) / 256, 256, 0, stream>>>(Wb[0], WbT[0], 64, 64);
    w_to_bf16T<<<(64 * 64 + 255) / 256, 256, 0, stream>>>(Wb[1], WbT[1], 64, 64);
    w_to_bf16T<<<(64 * 32 + 255) / 256, 256, 0, stream>>>(Wb[2], WbT[2], 64, 32);

    dim3 ggrid(GIN_N / 64, GIN_B);                       // 64 x 4 blocks, 8 waves each
    int  mlp_blocks = (GIN_B * GIN_N * GIN_K / 16) / 8;  // 256

    // ---- Layer 0 (CIN=32, KC=64) ----
    gin_prep<<<(GIN_B * GIN_N * 64) / 256, 256, 0, stream>>>(
        x, mask, ba[0], ba[0], ba[0], ba[0], 32, 0, xf, xfT);
    gin_adj_gemm<64><<<ggrid, 256, 0, stream>>>(adj, xfT, xf, agg);
    gin_mlp<32, 64><<<mlp_blocks, 256, 0, stream>>>(
        agg, WaT[0], ba[0], bng[0], bnb[0], bnm[0], bnv[0], WbT[0], bb[0], mask, xcur);

    // ---- Layer 1 (CIN=64, KC=128) ----
    gin_prep<<<(GIN_B * GIN_N * 128) / 256, 256, 0, stream>>>(
        xcur, mask, og[0], ob[0], om[0], ov[0], 64, 1, xf, xfT);
    gin_adj_gemm<128><<<ggrid, 256, 0, stream>>>(adj, xfT, xf, agg);
    gin_mlp<64, 64><<<mlp_blocks, 256, 0, stream>>>(
        agg, WaT[1], ba[1], bng[1], bnb[1], bnm[1], bnv[1], WbT[1], bb[1], mask, xcur);

    // ---- Layer 2 (CIN=64, KC=128, COUT=32 -> d_out) ----
    gin_prep<<<(GIN_B * GIN_N * 128) / 256, 256, 0, stream>>>(
        xcur, mask, og[1], ob[1], om[1], ov[1], 64, 1, xf, xfT);
    gin_adj_gemm<128><<<ggrid, 256, 0, stream>>>(adj, xfT, xf, agg);
    gin_mlp<64, 32><<<mlp_blocks, 256, 0, stream>>>(
        agg, WaT[2], ba[2], bng[2], bnb[2], bnm[2], bnv[2], WbT[2], bb[2], mask, (float*)d_out);
}
